// SRCNN_CAB_39633958207681
// MI455X (gfx1250) — matrix operand order repaired and checked by reading, not compiled, not run on hardware
//
#include <hip/hip_runtime.h>
#include <hip/hip_bf16.h>

typedef __attribute__((ext_vector_type(16))) _Float16 v16h;
typedef __attribute__((ext_vector_type(8)))  _Float16 v8h;
typedef __attribute__((ext_vector_type(8)))  float    v8f;
typedef __attribute__((ext_vector_type(4)))  float    v4f;

#define BN   16
#define HU   256
#define WU   256
#define CH   64
#define FCN  2048
#define N1   5184   // 64*81
#define LL2  225    // 15*15

__device__ __forceinline__ v8f wmma16(v16h a, v16h b, v8f c) {
  return __builtin_amdgcn_wmma_f32_16x16x32_f16(
      /*neg_a=*/false, a, /*neg_b=*/false, b,
      /*c_mod=*/(short)0, c, /*reuse_a=*/false, /*reuse_b=*/false);
}

// ---------------------------------------------------------------- K1: kfc
// kfc[b][n] = relu(kernel[b]·kl_w[n] + kl_b[n]) stored f16  (16x2048)
__global__ void k_kfc(const float* __restrict__ ker, const float* __restrict__ klw,
                      const float* __restrict__ klb, _Float16* __restrict__ kfc16) {
  int idx = blockIdx.x * 256 + threadIdx.x;          // 32768
  int b = idx >> 11, n = idx & 2047;
  float acc = klb[n];
  const float* kp = ker + b * LL2;
  const float* wp = klw + (size_t)n * LL2;
  for (int i = 0; i < LL2; ++i) acc += kp[i] * wp[i];
  acc = acc > 0.f ? acc : 0.f;
  kfc16[idx] = (_Float16)acc;
}

// ---------------------------------------------------------------- K2: w1 GEMM (WMMA)
// w1h[b][n] = kfc[b]·w1_w[n] + w1_b[n]   M=16(batch) N=5184 K=2048, f32->f16 B on the fly
__global__ void k_w1(const _Float16* __restrict__ kfc16, const float* __restrict__ w1w,
                     const float* __restrict__ w1b, _Float16* __restrict__ w1h) {
  int lane = threadIdx.x & 31;
  int row  = lane & 15;          // A row (batch) / B-D col
  int hsel = lane >> 4;
  int kbA  = hsel * 16;
  int kbB  = hsel * 16;
  int n0   = blockIdx.x * 16;    // 324 blocks
  int n    = n0 + row;

  v8f acc;
  float bias = w1b[n];
  #pragma unroll
  for (int r = 0; r < 8; ++r) acc[r] = bias;

  const _Float16* ap = kfc16 + (size_t)row * FCN;
  const float*    bp = w1w + (size_t)n * FCN;
  for (int k = 0; k < FCN; k += 32) {
    v8h alo = *(const v8h*)(ap + k + kbA);
    v8h ahi = *(const v8h*)(ap + k + kbA + 8);
    v16h a = __builtin_shufflevector(alo, ahi, 0,1,2,3,4,5,6,7,8,9,10,11,12,13,14,15);
    v16h bb;
    const float* bk = bp + k + kbB;
    #pragma unroll
    for (int q = 0; q < 4; ++q) {
      v4f f = *(const v4f*)(bk + q * 4);
      bb[q*4+0] = (_Float16)f[0]; bb[q*4+1] = (_Float16)f[1];
      bb[q*4+2] = (_Float16)f[2]; bb[q*4+3] = (_Float16)f[3];
    }
    acc = wmma16(a, bb, acc);
  }
  #pragma unroll
  for (int r = 0; r < 8; ++r) {
    int m = hsel * 8 + r;                       // batch
    w1h[(size_t)m * N1 + n] = (_Float16)acc[r];
  }
}

// ---------------------------------------------------------------- K3: bilinear x2 upsample -> f16
__global__ void k_upsample(const float* __restrict__ img, _Float16* __restrict__ up) {
  int idx = blockIdx.x * 256 + threadIdx.x;          // 1M
  int b = idx >> 16, y = (idx >> 8) & 255, x = idx & 255;
  float sy = 0.5f * y - 0.25f, sx = 0.5f * x - 0.25f;
  int y0 = (int)floorf(sy), x0 = (int)floorf(sx);
  float wy = sy - y0, wx = sx - x0;
  int y0c = min(max(y0, 0), 127), y1c = min(max(y0 + 1, 0), 127);
  int x0c = min(max(x0, 0), 127), x1c = min(max(x0 + 1, 0), 127);
  const float* p = img + (size_t)b * 128 * 128;
  float v = (1.f - wy) * ((1.f - wx) * p[y0c * 128 + x0c] + wx * p[y0c * 128 + x1c])
          +        wy  * ((1.f - wx) * p[y1c * 128 + x0c] + wx * p[y1c * 128 + x1c]);
  up[idx] = (_Float16)v;
}

// ---------------------------------------------------------------- K4: pack c2_w -> Bc[oc][576] f16
__global__ void k_c2pack(const float* __restrict__ c2w, _Float16* __restrict__ bc) {
  int i = blockIdx.x * 256 + threadIdx.x;            // 36864
  int oc = i / 576, r = i % 576, tap = r >> 6, ic = r & 63;
  bc[i] = (_Float16)c2w[((size_t)(oc * 64 + ic)) * 9 + tap];
}

// ---------------------------------------------------------------- K5: conv1 (dynamic 9x9) WMMA
// f1[b,y,x,oc] = relu(sum_tap up[b,y+dy-4,x+dx-4] * w1h[b, oc*81+tap]); K=81 pad 96
__global__ void k_conv1(const _Float16* __restrict__ up, const _Float16* __restrict__ w1h,
                        _Float16* __restrict__ f1) {
  __shared__ __align__(32) _Float16 Bc1[64 * 96];
  __shared__ __align__(32) _Float16 win[9 * 136];
  int tid  = threadIdx.x;
  int wave = tid >> 5, lane = tid & 31;
  int xhalf = blockIdx.x & 1;
  int y     = (blockIdx.x >> 1) & 255;
  int b     = blockIdx.x >> 9;                        // 8192 blocks
  int x0blk = xhalf * 128;

  for (int i = tid; i < 64 * 96; i += 256) {
    int oc = i / 96, t = i % 96;
    Bc1[i] = (t < 81) ? w1h[(size_t)b * N1 + oc * 81 + t] : (_Float16)0;
  }
  for (int i = tid; i < 9 * 136; i += 256) {
    int ry = i / 136, rx = i % 136;
    int yy = y + ry - 4, xx = x0blk + rx - 4;
    _Float16 v = (_Float16)0;
    if (yy >= 0 && yy < HU && xx >= 0 && xx < WU)
      v = up[((size_t)b * HU + yy) * WU + xx];
    win[i] = v;
  }
  __syncthreads();

  int row = lane & 15, hsel = lane >> 4;
  int kbA = hsel * 8, kbB = hsel * 16;
  int x0  = x0blk + wave * 16;
  int xb  = wave * 16 + row;

  v8f acc[4];
  #pragma unroll
  for (int t = 0; t < 4; ++t)
    #pragma unroll
    for (int r = 0; r < 8; ++r) acc[t][r] = 0.f;

  for (int ks = 0; ks < 3; ++ks) {
    v16h a;
    #pragma unroll
    for (int j = 0; j < 16; ++j) {
      int k = ks * 32 + kbB + j;
      _Float16 v = (_Float16)0;
      if (k < 81) { int ty = k / 9, tx = k - ty * 9; v = win[ty * 136 + xb + tx]; }
      a[j] = v;
    }
    #pragma unroll
    for (int t = 0; t < 4; ++t) {
      v16h bb = *(const v16h*)&Bc1[(t * 16 + row) * 96 + ks * 32 + kbB];
      acc[t] = wmma16(a, bb, acc[t]);
    }
  }
  _Float16* dst = f1 + (((size_t)b * HU + y) * WU + x0) * CH;
  #pragma unroll
  for (int t = 0; t < 4; ++t)
    #pragma unroll
    for (int r = 0; r < 8; ++r) {
      int m = hsel * 8 + r;
      float v = acc[t][r]; v = v > 0.f ? v : 0.f;
      dst[(size_t)m * CH + t * 16 + row] = (_Float16)v;
    }
}

// ---------------------------------------------------------------- K6: conv2 64->64 3x3 WMMA (dominant)
// K = 9 taps * 64 ic = 576 = 18 ksteps; A direct from f1 NHWC f16, B from packed Bc
__global__ void k_conv2(const _Float16* __restrict__ f1, const _Float16* __restrict__ bc,
                        const float* __restrict__ c2b, _Float16* __restrict__ f2) {
  int tid = threadIdx.x;
  int wave = tid >> 5, lane = tid & 31;
  int xhalf = blockIdx.x & 1;
  int y     = (blockIdx.x >> 1) & 255;
  int b     = blockIdx.x >> 9;                        // 8192 blocks
  int x0    = xhalf * 128 + wave * 16;
  int row = lane & 15, hsel = lane >> 4;
  int kbA = hsel * 16, kbB = hsel * 16;

  v8f acc[4];
  #pragma unroll
  for (int t = 0; t < 4; ++t) {
    float bias = c2b[t * 16 + row];
    #pragma unroll
    for (int r = 0; r < 8; ++r) acc[t][r] = bias;
  }

  #pragma unroll
  for (int tap = 0; tap < 9; ++tap) {
    int dy = tap / 3 - 1, dx = tap % 3 - 1;
    int yy = y + dy, xx = x0 + row + dx;
    bool inb = (yy >= 0 && yy < HU && xx >= 0 && xx < WU);
    const _Float16* src = f1 + (((size_t)b * HU + (inb ? yy : 0)) * WU + (inb ? xx : 0)) * CH;
    #pragma unroll
    for (int ks = 0; ks < 2; ++ks) {
      v16h a = {};
      if (inb) {
        v8h lo = *(const v8h*)(src + ks * 32 + kbA);
        v8h hi = *(const v8h*)(src + ks * 32 + kbA + 8);
        a = __builtin_shufflevector(lo, hi, 0,1,2,3,4,5,6,7,8,9,10,11,12,13,14,15);
      }
      int k0 = tap * 64 + ks * 32;
      #pragma unroll
      for (int t = 0; t < 4; ++t) {
        v16h bb = *(const v16h*)(bc + (size_t)(t * 16 + row) * 576 + k0 + kbB);
        acc[t] = wmma16(a, bb, acc[t]);
      }
    }
  }
  _Float16* dst = f2 + (((size_t)b * HU + y) * WU + x0) * CH;
  #pragma unroll
  for (int t = 0; t < 4; ++t)
    #pragma unroll
    for (int r = 0; r < 8; ++r) {
      int m = hsel * 8 + r;
      float v = acc[t][r]; v = v > 0.f ? v : 0.f;
      dst[(size_t)m * CH + t * 16 + row] = (_Float16)v;
    }
}

// ---------------------------------------------------------------- K7: conv3 64->1 5x5 (VALU, N=1)
__global__ void k_conv3(const _Float16* __restrict__ f2, const float* __restrict__ c3w,
                        const float* __restrict__ c3b, float* __restrict__ out) {
  __shared__ float c3s[64 * 25];
  int tid = threadIdx.x;
  for (int i = tid; i < 1600; i += 256) c3s[i] = c3w[i];
  __syncthreads();
  int idx = blockIdx.x * 256 + tid;                  // 1M
  int b = idx >> 16, y = (idx >> 8) & 255, x = idx & 255;
  float acc = c3b[0];
  for (int dy = 0; dy < 5; ++dy) {
    int yy = y + dy - 2;
    if (yy < 0 || yy >= HU) continue;
    for (int dx = 0; dx < 5; ++dx) {
      int xx = x + dx - 2;
      if (xx < 0 || xx >= WU) continue;
      const _Float16* p = f2 + (((size_t)b * HU + yy) * WU + xx) * CH;
      int tap = dy * 5 + dx;
      #pragma unroll
      for (int oc8 = 0; oc8 < 8; ++oc8) {
        v8h v = *(const v8h*)(p + oc8 * 8);
        #pragma unroll
        for (int j = 0; j < 8; ++j)
          acc += (float)v[j] * c3s[(oc8 * 8 + j) * 25 + tap];
      }
    }
  }
  out[idx] = acc;
}

// ---------------------------------------------------------------- launch
extern "C" void kernel_launch(void* const* d_in, const int* in_sizes, int n_in,
                              void* d_out, int out_size, void* d_ws, size_t ws_size,
                              hipStream_t stream) {
  const float* img  = (const float*)d_in[0];
  const float* ker  = (const float*)d_in[1];
  const float* klw  = (const float*)d_in[2];
  const float* klb  = (const float*)d_in[3];
  const float* w1w  = (const float*)d_in[4];
  const float* w1b  = (const float*)d_in[5];
  const float* c2w  = (const float*)d_in[6];
  const float* c2b  = (const float*)d_in[7];
  const float* c3w  = (const float*)d_in[8];
  const float* c3b  = (const float*)d_in[9];
  float* out = (float*)d_out;

  char* ws = (char*)d_ws;
  // 32B-aligned scratch layout (~271 MB total)
  const size_t OFF_KFC  = 0;                      // 16*2048*2      =     65536
  const size_t OFF_W1H  = 65536;                  // 16*5184*2      =    165888
  const size_t OFF_C2W  = 231424;                 // 64*576*2       =     73728
  const size_t OFF_UP   = 305152;                 // 16*256*256*2   =   2097152
  const size_t OFF_F1   = 2402304;                // 16*256*256*64*2= 134217728
  const size_t OFF_F2   = 136620032;              // same
  _Float16* kfc16 = (_Float16*)(ws + OFF_KFC);
  _Float16* w1h   = (_Float16*)(ws + OFF_W1H);
  _Float16* c2wh  = (_Float16*)(ws + OFF_C2W);
  _Float16* up    = (_Float16*)(ws + OFF_UP);
  _Float16* f1    = (_Float16*)(ws + OFF_F1);
  _Float16* f2    = (_Float16*)(ws + OFF_F2);

  k_kfc     <<<128, 256, 0, stream>>>(ker, klw, klb, kfc16);
  k_w1      <<<324,  32, 0, stream>>>(kfc16, w1w, w1b, w1h);
  k_upsample<<<4096, 256, 0, stream>>>(img, up);
  k_c2pack  <<<144, 256, 0, stream>>>(c2w, c2wh);
  k_conv1   <<<8192, 256, 0, stream>>>(up, w1h, f1);
  k_conv2   <<<8192, 256, 0, stream>>>(f1, c2wh, c2b, f2);
  k_conv3   <<<4096, 256, 0, stream>>>(f2, c3w, c3b, out);
}
